// MiniGPTModel_89687507076427
// MI455X (gfx1250) — matrix operand order repaired and checked by reading, not compiled, not run on hardware
//
#include <hip/hip_runtime.h>

// MiniGPT fused attention forward for gfx1250 (CDNA5, wave32, WMMA).
//   B=8, T=2048, E=256.  All matmuls via v_wmma_f32_16x16x32_f16.
//   Pipeline: [convert W f32->f16] -> [embed-gather + QKV proj, WMMA]
//             -> [flash attention: QK^T WMMA, online softmax, PV WMMA].
//   Never materializes the 134MB score tensor (saves ~1GB of HBM traffic).

#define BB 8
#define TT 2048
#define EE 256

typedef __attribute__((ext_vector_type(16))) _Float16 v16h;
typedef __attribute__((ext_vector_type(8)))  float    v8f;

union HVec { v16h v; _Float16 h[16]; unsigned int u[8]; };

__device__ __forceinline__ v8f wmma_f16(v16h a, v16h b, v8f c) {
  // 8 args: (neg_a, A, neg_b, B, c_mod, C, reuse_a, reuse_b)
  return __builtin_amdgcn_wmma_f32_16x16x32_f16(false, a, false, b, (short)0, c,
                                                false, false);
}

// ---------------------------------------------------------------- kernel 1
__global__ __launch_bounds__(256) void w_to_f16(const float* __restrict__ w,
                                                _Float16* __restrict__ o, int n) {
  int i = blockIdx.x * blockDim.x + threadIdx.x;
  if (i < n) o[i] = (_Float16)w[i];
}

// ---------------------------------------------------------------- kernel 2
// One wave per (batch, 16-row t-tile).  h = tok_emb[x] + pos_emb gathered
// straight into the WMMA A layout; q,k stored row-major f16, v stored
// transposed [B,E,T] so the PV B-operand later loads contiguous dwords.
__global__ __launch_bounds__(256, 1) void qkv_wmma(
    const int*   __restrict__ x,  const float* __restrict__ tok,
    const float* __restrict__ pos,
    const _Float16* __restrict__ wq, const _Float16* __restrict__ wk,
    const _Float16* __restrict__ wv,
    const float* __restrict__ bq, const float* __restrict__ bk,
    const float* __restrict__ bv,
    _Float16* __restrict__ qh, _Float16* __restrict__ kh,
    _Float16* __restrict__ vT) {
  const int lane  = threadIdx.x & 31;
  const int wave  = threadIdx.x >> 5;
  const int tile  = blockIdx.x * 8 + wave;     // 0..1023
  const int b     = tile >> 7;                 // 128 t-tiles per batch
  const int tbase = (tile & 127) << 4;
  const int half  = lane >> 4;
  const int row   = lane & 15;

  // Resident A chunks: h[t, e] in f16, 8 chunks of 16x32 covering E=256.
  const int t_a = tbase + row;
  const int xid = x[b * TT + t_a];
  const float* te = tok + (unsigned)xid * EE;
  const float* pe = pos + (unsigned)t_a * EE;
  HVec ha[8];
#pragma unroll
  for (int c = 0; c < 8; ++c) {
#pragma unroll
    for (int v = 0; v < 8; ++v) {
      const int e0 = 32 * c + 16 * (v >> 2) + 8 * half + 2 * (v & 3);
      ha[c].h[2 * v]     = (_Float16)(te[e0]     + pe[e0]);
      ha[c].h[2 * v + 1] = (_Float16)(te[e0 + 1] + pe[e0 + 1]);
    }
  }

  const int n = row;  // output feature column within a 16-wide N tile
#pragma unroll
  for (int proj = 0; proj < 3; ++proj) {
    const _Float16* wh = (proj == 0) ? wq : (proj == 1) ? wk : wv;
    const float*    bb = (proj == 0) ? bq : (proj == 1) ? bk : bv;
    for (int nt = 0; nt < 16; ++nt) {
      const int f = nt * 16 + n;
      const float bias = bb[f];
      v8f acc = {bias, bias, bias, bias, bias, bias, bias, bias};
#pragma unroll
      for (int c = 0; c < 8; ++c) {
        HVec wb;  // B(e, f) = W[f*E + e]; packed pair along e -> dword load
#pragma unroll
        for (int v = 0; v < 8; ++v) {
          const int e0 = 32 * c + 16 * (v >> 2) + 8 * half + 2 * (v & 3);
          wb.u[v] = *(const unsigned int*)(wh + f * EE + e0);
        }
        acc = wmma_f16(ha[c].v, wb.v, acc);
      }
#pragma unroll
      for (int r = 0; r < 8; ++r) {
        const int tm = tbase + r + 8 * half;           // C/D row
        const _Float16 val = (_Float16)acc[r];
        if (proj == 0)      qh[(b * TT + tm) * EE + f] = val;
        else if (proj == 1) kh[(b * TT + tm) * EE + f] = val;
        else                vT[(b * EE + f) * TT + tm] = val;   // transposed
      }
    }
  }
}

// ---------------------------------------------------------------- kernel 3
// Flash attention: one wave per 16-query tile, 32 keys per iteration.
// Score tiles land in C/D layout; P is relayouted to A via a per-wave LDS slab.
__global__ __launch_bounds__(256, 1) void attn_wmma(
    const _Float16* __restrict__ qh, const _Float16* __restrict__ kh,
    const _Float16* __restrict__ vT, float* __restrict__ out) {
  __shared__ _Float16 lds_p[8][16 * 32];   // 1KB per wave
  const int lane  = threadIdx.x & 31;
  const int wave  = threadIdx.x >> 5;
  const int tile  = blockIdx.x * 8 + wave;
  const int b     = tile >> 7;
  const int qbase = (tile & 127) << 4;
  const int half  = lane >> 4;
  const int row   = lane & 15;
  const int n     = row;

  // Resident Q in A layout (8 chunks of 16x32).
  HVec qa[8];
#pragma unroll
  for (int c = 0; c < 8; ++c)
#pragma unroll
    for (int v = 0; v < 8; ++v) {
      const int e0 = 32 * c + 16 * (v >> 2) + 8 * half + 2 * (v & 3);
      qa[c].u[v] =
          *(const unsigned int*)(qh + (b * TT + qbase + row) * EE + e0);
    }

  v8f O[16];
#pragma unroll
  for (int nt = 0; nt < 16; ++nt) O[nt] = (v8f){0, 0, 0, 0, 0, 0, 0, 0};
  float m_run[8], l_run[8];
#pragma unroll
  for (int r = 0; r < 8; ++r) { m_run[r] = -1e30f; l_run[r] = 0.0f; }

  volatile _Float16* pbuf = lds_p[wave];

  for (int kb = 0; kb <= qbase + 15; kb += 32) {
    // ---- S = Q K^T for 32 keys (two 16x16 C tiles), f32 accumulate
    v8f s0 = {0, 0, 0, 0, 0, 0, 0, 0};
    v8f s1 = {0, 0, 0, 0, 0, 0, 0, 0};
    const int tk0 = kb + n;
    const int tk1 = kb + 16 + n;
#pragma unroll
    for (int c = 0; c < 8; ++c) {
      HVec kb0, kb1;  // B(e, tk) = k[tk*E + e]; pair along e -> dword load
#pragma unroll
      for (int v = 0; v < 8; ++v) {
        const int e0 = 32 * c + 16 * (v >> 2) + 8 * half + 2 * (v & 3);
        kb0.u[v] = *(const unsigned int*)(kh + (b * TT + tk0) * EE + e0);
        kb1.u[v] = *(const unsigned int*)(kh + (b * TT + tk1) * EE + e0);
      }
      s0 = wmma_f16(qa[c].v, kb0.v, s0);
      s1 = wmma_f16(qa[c].v, kb1.v, s1);
    }

    // ---- causal mask + online softmax (per C/D row, reduce over 16 lanes)
    float p0[8], p1[8], scale[8];
#pragma unroll
    for (int r = 0; r < 8; ++r) {
      const int qg = qbase + r + 8 * half;
      const float v0 = (tk0 <= qg) ? s0[r] : -1e30f;
      const float v1 = (tk1 <= qg) ? s1[r] : -1e30f;
      float mx = fmaxf(v0, v1);
      mx = fmaxf(mx, __shfl_xor(mx, 1));
      mx = fmaxf(mx, __shfl_xor(mx, 2));
      mx = fmaxf(mx, __shfl_xor(mx, 4));
      mx = fmaxf(mx, __shfl_xor(mx, 8));
      const float m_new = fmaxf(m_run[r], mx);
      scale[r] = __expf(m_run[r] - m_new);
      m_run[r] = m_new;
      p0[r] = __expf(v0 - m_new);
      p1[r] = __expf(v1 - m_new);
      float rs = p0[r] + p1[r];
      rs += __shfl_xor(rs, 1);
      rs += __shfl_xor(rs, 2);
      rs += __shfl_xor(rs, 4);
      rs += __shfl_xor(rs, 8);
      l_run[r] = l_run[r] * scale[r] + rs;
    }
#pragma unroll
    for (int nt = 0; nt < 16; ++nt)
#pragma unroll
      for (int r = 0; r < 8; ++r) O[nt][r] *= scale[r];

    // ---- relayout P (C/D -> A) via per-wave LDS slab, f16
#pragma unroll
    for (int r = 0; r < 8; ++r) {
      const int m = r + 8 * half;
      pbuf[m * 32 + n]      = (_Float16)p0[r];
      pbuf[m * 32 + 16 + n] = (_Float16)p1[r];
    }
    __builtin_amdgcn_wave_barrier();   // same-wave LDS ops are in order
    HVec pa;
#pragma unroll
    for (int v = 0; v < 8; ++v) {
      const int kk = 16 * (v >> 2) + 8 * half + 2 * (v & 3);
      pa.u[v] = *(volatile const unsigned int*)(pbuf + row * 32 + kk);
    }
    __builtin_amdgcn_wave_barrier();

    // ---- O += P x V ; B(tk, e) = vT[(b*E+e)*T + tk], pair along tk -> dword
#pragma unroll
    for (int nt = 0; nt < 16; ++nt) {
      HVec vb;
      const int e = nt * 16 + n;
#pragma unroll
      for (int v = 0; v < 8; ++v) {
        const int k0 = 16 * (v >> 2) + 8 * half + 2 * (v & 3);
        vb.u[v] = *(const unsigned int*)(vT + (b * EE + e) * TT + kb + k0);
      }
      O[nt] = wmma_f16(pa.v, vb.v, O[nt]);
    }
  }

  // ---- epilogue: normalize and store f32
  float inv_l[8];
#pragma unroll
  for (int r = 0; r < 8; ++r) inv_l[r] = 1.0f / l_run[r];
#pragma unroll
  for (int nt = 0; nt < 16; ++nt)
#pragma unroll
    for (int r = 0; r < 8; ++r) {
      const int tm = qbase + r + 8 * half;
      out[(b * TT + tm) * EE + nt * 16 + n] = O[nt][r] * inv_l[r];
    }
}

// ---------------------------------------------------------------- launcher
extern "C" void kernel_launch(void* const* d_in, const int* in_sizes, int n_in,
                              void* d_out, int out_size, void* d_ws,
                              size_t ws_size, hipStream_t stream) {
  (void)in_sizes; (void)n_in; (void)out_size; (void)ws_size;
  const int*   x   = (const int*)d_in[0];
  const float* tok = (const float*)d_in[1];
  const float* pos = (const float*)d_in[2];
  const float* Wq  = (const float*)d_in[3];
  const float* bq  = (const float*)d_in[4];
  const float* Wk  = (const float*)d_in[5];
  const float* bk  = (const float*)d_in[6];
  const float* Wv  = (const float*)d_in[7];
  const float* bv  = (const float*)d_in[8];
  float* out = (float*)d_out;

  // Workspace layout (f16): q[BTE] k[BTE] vT[B,E,T] wq wk wv  (~24.4MB)
  _Float16* ws  = (_Float16*)d_ws;
  const int BTE = BB * TT * EE;       // 4,194,304
  const int WSZ = EE * EE;            // 65,536
  _Float16* qh  = ws;
  _Float16* kh  = qh + BTE;
  _Float16* vT  = kh + BTE;
  _Float16* wqh = vT + BTE;
  _Float16* wkh = wqh + WSZ;
  _Float16* wvh = wkh + WSZ;

  w_to_f16<<<WSZ / 256, 256, 0, stream>>>(Wq, wqh, WSZ);
  w_to_f16<<<WSZ / 256, 256, 0, stream>>>(Wk, wkh, WSZ);
  w_to_f16<<<WSZ / 256, 256, 0, stream>>>(Wv, wvh, WSZ);

  // 1024 (b, t-tile) waves, 8 waves (256 threads) per block
  qkv_wmma<<<128, 256, 0, stream>>>(x, tok, pos, wqh, wkh, wvh, bq, bk, bv,
                                    qh, kh, vT);
  attn_wmma<<<128, 256, 0, stream>>>(qh, kh, vT, out);
}
